// AttentionSE3_53523882442966
// MI455X (gfx1250) — matrix operand order repaired and checked by reading, not compiled, not run on hardware
//
#include <hip/hip_runtime.h>
#include <math.h>

// ---------------------------------------------------------------------------
// SE(3) attention forward for MI455X (gfx1250, wave32, WMMA).
// Dominant compute = per-edge radial-MLP final layer: (4096 x 128) @ (128 x
// 49152) ~= 51.5 GFLOP, fused with the per-edge basis contraction so the
// 805MB "R" tensor is never materialized.  bf16 WMMA 16x16x32, f32 accum.
// Workspace usage: ~48 MB (h bf16 4.2MB | w3^T bf16 12.6MB | v f32 33.5MB).
// ---------------------------------------------------------------------------

typedef __attribute__((ext_vector_type(8)))  __bf16 v8bf;
typedef __attribute__((ext_vector_type(16))) __bf16 v16bf;
typedef __attribute__((ext_vector_type(8)))  float  v8f;

#define N_NODES 256
#define J_NB    16
#define CIN     32
#define HID     256
#define MIDW    128
#define EDGES   (N_NODES * J_NB)
#define LN_EPS  1e-5f

__device__ __forceinline__ __bf16 f2bf(float f) {
  union { float f; unsigned u; } v; v.f = f;
  unsigned r = v.u + 0x7FFFu + ((v.u >> 16) & 1u);   // round-to-nearest-even
  unsigned short h = (unsigned short)(r >> 16);
  __bf16 o; __builtin_memcpy(&o, &h, 2); return o;
}

__device__ __forceinline__ float gelu_exact(float x) {
  return 0.5f * x * (1.0f + erff(x * 0.70710678118654752f));
}

// ---------------------------------------------------------------------------
// Kernel 1: radial MLP layers 1+2 (LN + exact GELU), per (edge, degree-pair).
// Output: h bf16, layout [dp][edge][k] (k contiguous -> WMMA-B friendly).
// ---------------------------------------------------------------------------
struct RPPtrs { const float *w1, *b1, *g1, *bb1, *w2, *b2, *g2, *bb2; };
struct RP4 { RPPtrs p[4]; };

__global__ __launch_bounds__(128) void radial12_kernel(
    const float* __restrict__ rel_dist, RP4 rp, __bf16* __restrict__ hws) {
  __shared__ float sred[128];
  __shared__ float h1s[128];
  const int k    = threadIdx.x;
  const int edge = blockIdx.x;
  const int dp   = blockIdx.y;
  const RPPtrs P = rp.p[dp];

  const float dist = rel_dist[edge];

  // ---- layer 1: scalar -> 128, LN, GELU ----
  float x = dist * P.w1[k] + P.b1[k];
  sred[k] = x; __syncthreads();
  for (int s = 64; s > 0; s >>= 1) { if (k < s) sred[k] += sred[k + s]; __syncthreads(); }
  float mu = sred[0] * (1.0f / 128.0f); __syncthreads();
  float d  = x - mu;
  sred[k] = d * d; __syncthreads();
  for (int s = 64; s > 0; s >>= 1) { if (k < s) sred[k] += sred[k + s]; __syncthreads(); }
  float var = sred[0] * (1.0f / 128.0f); __syncthreads();
  float y = d * rsqrtf(var + LN_EPS) * P.g1[k] + P.bb1[k];
  h1s[k] = gelu_exact(y);
  __syncthreads();

  // ---- layer 2: 128 -> 128, LN, GELU ----
  float acc = P.b2[k];
  #pragma unroll 8
  for (int j = 0; j < 128; ++j) acc += h1s[j] * P.w2[j * MIDW + k];
  __syncthreads();
  sred[k] = acc; __syncthreads();
  for (int s = 64; s > 0; s >>= 1) { if (k < s) sred[k] += sred[k + s]; __syncthreads(); }
  mu = sred[0] * (1.0f / 128.0f); __syncthreads();
  d  = acc - mu;
  sred[k] = d * d; __syncthreads();
  for (int s = 64; s > 0; s >>= 1) { if (k < s) sred[k] += sred[k + s]; __syncthreads(); }
  var = sred[0] * (1.0f / 128.0f); __syncthreads();
  y = d * rsqrtf(var + LN_EPS) * P.g2[k] + P.bb2[k];
  float h2 = gelu_exact(y);

  hws[((size_t)dp * EDGES + edge) * MIDW + k] = f2bf(h2);
}

// ---------------------------------------------------------------------------
// Kernel 2: w3 (f32, [k][m]) -> w3^T (bf16, [m][k])  (A-matrix layout).
// One-time 25MB read; stays hot in 192MB L2 afterwards.
// ---------------------------------------------------------------------------
__global__ __launch_bounds__(256) void w3cvt_kernel(
    const float* __restrict__ w3, __bf16* __restrict__ w3t, int od) {
  long gid = (long)blockIdx.x * 256 + threadIdx.x;   // k-major for coalesced read
  long total = (long)od * MIDW;
  if (gid >= total) return;
  int k = (int)(gid / od);
  int m = (int)(gid % od);
  w3t[(size_t)m * MIDW + k] = f2bf(w3[(size_t)k * od + m]);
}

// ---------------------------------------------------------------------------
// Kernel 3: fused (h @ w3 + b3) GEMM + per-edge tmp contraction.
// Transposed GEMM: A = w3^T rows = out-cols (e,c,f), B = h^T cols = 16 edges.
// Block = 256 thr (8 waves) = one 16-edge tile x one degree pair; each wave
// owns 32 e-values.  Per (e, 16-col tile): 4x v_wmma_f32_16x16x32_bf16,
// bias add, per-lane serial (c,f) reduction with tmp coefficients, one
// shfl_xor(16) to merge row halves, register accumulation across tiles.
// ---------------------------------------------------------------------------
template <int F, int P, int IDIM>
__global__ __launch_bounds__(256) void edgegemm_kernel(
    const __bf16* __restrict__ hws,      // [edge][128] bf16 (dp-offset applied)
    const int*    __restrict__ nbr,      // [EDGES]
    const __bf16* __restrict__ w3t,      // [out_dim][128] bf16
    const float*  __restrict__ b3,       // [out_dim]
    const float*  __restrict__ basis,    // [EDGES][P][IDIM][F]
    const float*  __restrict__ xin,      // x0: [N][CIN] or x1: [N][CIN][3]
    float*        __restrict__ vout)     // [EDGES][HID][P]
{
  constexpr int COLS_E = CIN * F;        // columns of out_dim per e (mult of 16)
  constexpr int TILES  = COLS_E / 16;
  __shared__ float tmp_lds[COLS_E * P * 16];   // [colInE][p][edge16]

  const int tid   = threadIdx.x;
  const int ebase = blockIdx.x * 16;

  // ---- build tmp[c*F+fi][p][edge16] = sum_i x[nbr][c][i] * basis[p][i][fi] ----
  for (int pair = tid; pair < 16 * CIN; pair += 256) {
    const int et   = pair & 15;
    const int c    = pair >> 4;
    const int edge = ebase + et;
    const int nn   = nbr[edge];
    float xv[IDIM];
    #pragma unroll
    for (int ii = 0; ii < IDIM; ++ii)
      xv[ii] = (IDIM == 1) ? xin[nn * CIN + c] : xin[(nn * CIN + c) * 3 + ii];
    const float* bp = basis + (size_t)edge * P * IDIM * F;
    #pragma unroll
    for (int pp = 0; pp < P; ++pp)
      #pragma unroll
      for (int fi = 0; fi < F; ++fi) {
        float s = 0.0f;
        #pragma unroll
        for (int ii = 0; ii < IDIM; ++ii)
          s += xv[ii] * bp[(pp * IDIM + ii) * F + fi];
        tmp_lds[((c * F + fi) * P + pp) * 16 + et] = s;
      }
  }
  __syncthreads();

  const int lane  = tid & 31;
  const int n16   = lane & 15;       // edge within tile / B column
  const int khalf = lane >> 4;       // K-half select
  const int wave  = tid >> 5;
  const int edge  = ebase + n16;

  // ---- B-matrix registers (h^T), loaded once per wave ----
  v16bf B[4];
  {
    const __bf16* hp = hws + (size_t)edge * MIDW;
    #pragma unroll
    for (int kb = 0; kb < 4; ++kb)
      B[kb] = __builtin_shufflevector(*(const v8bf*)(hp + kb * 32 + khalf * 8),
          *(const v8bf*)(hp + kb * 32 + khalf * 8 + 16),
          0, 1, 2, 3, 4, 5, 6, 7, 8, 9, 10, 11, 12, 13, 14, 15);
  }

  for (int e = wave * 32; e < wave * 32 + 32; ++e) {
    float accp[P];
    #pragma unroll
    for (int pp = 0; pp < P; ++pp) accp[pp] = 0.0f;

    #pragma unroll
    for (int t = 0; t < TILES; ++t) {
      const int colbase = e * COLS_E + t * 16;
      const __bf16* ap = w3t + (size_t)(colbase + n16) * MIDW;

      v8f acc = {0.f, 0.f, 0.f, 0.f, 0.f, 0.f, 0.f, 0.f};
      #pragma unroll
      for (int kb = 0; kb < 4; ++kb) {
        v8bf lo = *(const v8bf*)(ap + kb * 32 + khalf * 8);
        v8bf hi = *(const v8bf*)(ap + kb * 32 + khalf * 8 + 16);
        v16bf A = __builtin_shufflevector(lo, hi,
            0, 1, 2, 3, 4, 5, 6, 7, 8, 9, 10, 11, 12, 13, 14, 15);
        acc = __builtin_amdgcn_wmma_f32_16x16x32_bf16(
            false, A, false, B[kb], (short)0, acc, false, false);
      }

      // bias + contraction: lane holds rows (khalf*8 + v) = 8 (c,f) indices
      const int rowbase = khalf * 8;
      const float* b3p  = b3 + colbase + rowbase;
      #pragma unroll
      for (int v = 0; v < 8; ++v) {
        const int colInE = t * 16 + rowbase + v;
        const float val  = acc[v] + b3p[v];
        #pragma unroll
        for (int pp = 0; pp < P; ++pp)
          accp[pp] += val * tmp_lds[(colInE * P + pp) * 16 + n16];
      }
    }

    // merge the two row halves; lanes 0-15 write v[edge][e][p]
    #pragma unroll
    for (int pp = 0; pp < P; ++pp) {
      float o = accp[pp] + __shfl_xor(accp[pp], 16, 32);
      if (khalf == 0)
        vout[((size_t)edge * HID + e) * P + pp] = o;
    }
  }
}

// ---------------------------------------------------------------------------
// Kernel 4: per-node attention (per degree) + output projection.  Tiny.
// ---------------------------------------------------------------------------
__global__ __launch_bounds__(256) void attn_kernel(
    const float* __restrict__ x0, const float* __restrict__ x1,
    const unsigned char* __restrict__ mask,
    const float* __restrict__ qw0, const float* __restrict__ qw1,
    const float* __restrict__ ow0, const float* __restrict__ ow1,
    const float* __restrict__ vA0, const float* __restrict__ vB0,  // p=1 pairs
    const float* __restrict__ vA1, const float* __restrict__ vB1,  // p=3 pairs
    float* __restrict__ out)
{
  __shared__ float v_lds[J_NB * HID * 3];   // 48 KB
  __shared__ float q_lds[HID * 3];
  __shared__ float o_lds[HID * 3];
  __shared__ float sim_lds[4 * 16];
  __shared__ float attn_lds[4 * 16];

  const int n   = blockIdx.x;
  const int tid = threadIdx.x;
  const float SCALE = 0.125f;               // 64^-0.5

  for (int deg = 0; deg < 2; ++deg) {
    const int pdim = (deg == 0) ? 1 : 3;
    const float* vA = (deg == 0) ? vA0 : vA1;
    const float* vB = (deg == 0) ? vB0 : vB1;
    const float* qw = (deg == 0) ? qw0 : qw1;
    const float* ow = (deg == 0) ? ow0 : ow1;
    const float* x  = (deg == 0) ? x0  : x1;

    // v[j][e][p] = sum of the two degree-pair contributions (contiguous slab)
    const int tot = J_NB * HID * pdim;
    const size_t gbase = (size_t)n * J_NB * HID * pdim;
    for (int i = tid; i < tot; i += 256)
      v_lds[i] = vA[gbase + i] + vB[gbase + i];

    // q[e][m]
    {
      const int e = tid;
      for (int m = 0; m < pdim; ++m) {
        float s = 0.0f;
        #pragma unroll 8
        for (int c = 0; c < CIN; ++c) {
          float xv = (pdim == 1) ? x[n * CIN + c] : x[(n * CIN + c) * 3 + m];
          s += xv * qw[c * HID + e];
        }
        q_lds[e * pdim + m] = s;
      }
    }
    __syncthreads();

    // sim[h][j]
    if (tid < 64) {
      const int h = tid >> 4, j = tid & 15;
      float s = 0.0f;
      for (int dh = 0; dh < 64; ++dh) {
        const int e = h * 64 + dh;
        for (int m = 0; m < pdim; ++m)
          s += q_lds[e * pdim + m] * v_lds[(j * HID + e) * pdim + m];
      }
      s *= SCALE;
      if (!mask[n * J_NB + j]) s = -3.4028234663852886e38f;
      sim_lds[tid] = s;
    }
    __syncthreads();

    // softmax over j (16) per head
    if (tid < 4) {
      float mx = -INFINITY;
      for (int j = 0; j < 16; ++j) mx = fmaxf(mx, sim_lds[tid * 16 + j]);
      float ss = 0.0f;
      for (int j = 0; j < 16; ++j) {
        float ev = __expf(sim_lds[tid * 16 + j] - mx);
        attn_lds[tid * 16 + j] = ev; ss += ev;
      }
      float inv = 1.0f / ss;
      for (int j = 0; j < 16; ++j) attn_lds[tid * 16 + j] *= inv;
    }
    __syncthreads();

    // o[e][m] = sum_j attn * v
    {
      const int e = tid, h = e >> 6;
      for (int m = 0; m < pdim; ++m) {
        float s = 0.0f;
        #pragma unroll
        for (int j = 0; j < 16; ++j)
          s += attn_lds[h * 16 + j] * v_lds[(j * HID + e) * pdim + m];
        o_lds[e * pdim + m] = s;
      }
    }
    __syncthreads();

    // output projection: out[c][m] = sum_e o[e][m] * ow[e][c]
    for (int i = tid; i < CIN * pdim; i += 256) {
      const int c = i / pdim, m = i % pdim;
      float s = 0.0f;
      #pragma unroll 8
      for (int e = 0; e < HID; ++e) s += o_lds[e * pdim + m] * ow[e * CIN + c];
      if (deg == 0) out[n * CIN + c] = s;
      else          out[8192 + (n * CIN + c) * 3 + m] = s;
    }
    __syncthreads();
  }
}

// ---------------------------------------------------------------------------
// Launch
// ---------------------------------------------------------------------------
extern "C" void kernel_launch(void* const* d_in, const int* in_sizes, int n_in,
                              void* d_out, int out_size, void* d_ws, size_t ws_size,
                              hipStream_t stream) {
  (void)in_sizes; (void)n_in; (void)out_size; (void)ws_size;

  const float* x0   = (const float*)d_in[0];
  const float* x1   = (const float*)d_in[1];
  const int*   nbr  = (const int*)  d_in[2];
  const float* rel  = (const float*)d_in[3];
  const float* basis[4] = { (const float*)d_in[4], (const float*)d_in[5],
                            (const float*)d_in[6], (const float*)d_in[7] };
  const float* qw0  = (const float*)d_in[8];
  const float* qw1  = (const float*)d_in[9];
  const float* ow0  = (const float*)d_in[10];
  const float* ow1  = (const float*)d_in[11];
  const unsigned char* mask = (const unsigned char*)d_in[12];

  RP4 rp;
  const float* w3[4]; const float* b3[4];
  for (int dp = 0; dp < 4; ++dp) {
    const int b = 13 + dp * 10;   // w1,b1,g1,bb1,w2,b2,g2,bb2,w3,b3
    rp.p[dp].w1  = (const float*)d_in[b + 0];
    rp.p[dp].b1  = (const float*)d_in[b + 1];
    rp.p[dp].g1  = (const float*)d_in[b + 2];
    rp.p[dp].bb1 = (const float*)d_in[b + 3];
    rp.p[dp].w2  = (const float*)d_in[b + 4];
    rp.p[dp].b2  = (const float*)d_in[b + 5];
    rp.p[dp].g2  = (const float*)d_in[b + 6];
    rp.p[dp].bb2 = (const float*)d_in[b + 7];
    w3[dp]       = (const float*)d_in[b + 8];
    b3[dp]       = (const float*)d_in[b + 9];
  }

  const int od[4] = { 8192, 8192, 8192, 24576 };   // out_dim per degree pair

  // workspace layout (bytes): h bf16 | w3^T bf16 | v f32   (~48 MB total)
  char* ws = (char*)d_ws;
  __bf16* hws = (__bf16*)ws;                              // 4*4096*128 bf16
  __bf16* w3t = (__bf16*)(ws + (size_t)4 * EDGES * MIDW * 2);
  const size_t w3toff[4] = { 0, 1048576, 2097152, 3145728 };
  float* vbuf = (float*)(ws + (size_t)4 * EDGES * MIDW * 2 + (size_t)6291456 * 2);
  const size_t voff[4] = { 0, 1048576, 4194304, 5242880 };

  // 1) radial MLP layers 1+2
  radial12_kernel<<<dim3(EDGES, 4), 128, 0, stream>>>(rel, rp, hws);

  // 2) w3 -> bf16 transposed (A layout)
  for (int dp = 0; dp < 4; ++dp) {
    long tot = (long)od[dp] * MIDW;
    w3cvt_kernel<<<(unsigned)((tot + 255) / 256), 256, 0, stream>>>(
        w3[dp], w3t + w3toff[dp], od[dp]);
  }

  // 3) fused WMMA GEMM + basis contraction, one instantiation per degree pair
  const dim3 gg(EDGES / 16);
  edgegemm_kernel<1, 1, 1><<<gg, 256, 0, stream>>>(     // '00'
      hws + (size_t)0 * EDGES * MIDW, nbr, w3t + w3toff[0], b3[0],
      basis[0], x0, vbuf + voff[0]);
  edgegemm_kernel<1, 3, 1><<<gg, 256, 0, stream>>>(     // '01'
      hws + (size_t)1 * EDGES * MIDW, nbr, w3t + w3toff[1], b3[1],
      basis[1], x0, vbuf + voff[1]);
  edgegemm_kernel<1, 1, 3><<<gg, 256, 0, stream>>>(     // '10'
      hws + (size_t)2 * EDGES * MIDW, nbr, w3t + w3toff[2], b3[2],
      basis[2], x1, vbuf + voff[2]);
  edgegemm_kernel<3, 3, 3><<<gg, 256, 0, stream>>>(     // '11'
      hws + (size_t)3 * EDGES * MIDW, nbr, w3t + w3toff[3], b3[3],
      basis[3], x1, vbuf + voff[3]);

  // 4) attention + output projection
  attn_kernel<<<N_NODES, 256, 0, stream>>>(
      x0, x1, mask, qw0, qw1, ow0, ow1,
      vbuf + voff[0], vbuf + voff[2],     // degree-0 contributions ('00' + '10')
      vbuf + voff[1], vbuf + voff[3],     // degree-1 contributions ('01' + '11')
      (float*)d_out);
}
